// GNNLayer_12670153523366
// MI455X (gfx1250) — compile-verified, matrix operand order checked by reading, not run on hardware
//
#include <hip/hip_runtime.h>
#include <hip/hip_bf16.h>
#include <math.h>

// ---------------- CDNA5 WMMA fragment types (wave32) ----------------
typedef __attribute__((ext_vector_type(16))) __bf16 v16bf; // 16x32 bf16 A/B frag (8 VGPRs)
typedef __attribute__((ext_vector_type(8)))  float  v8f;   // 16x16 f32 C/D (8 VGPRs)
typedef __attribute__((ext_vector_type(2)))  float  v2f;   // 16x4 f32 A / 4x16 f32 B frag (2 VGPRs)

// ---------------- problem constants ----------------
constexpr int kNRel  = 200;
constexpr int kVocab = 2 * kNRel + 1;  // 401
constexpr int kInDim = 128;
constexpr int kAttn  = 64;
constexpr int kDPath = 64;
constexpr int kNNode = 250000;
constexpr int kNEdge = 500000;
constexpr int kBatch = 64;

// ---------------- workspace layout (floats; ~10.3 MB total) ----------------
constexpr size_t WS_REL_ATTN = 0;                                  // vocab x 64
constexpr size_t WS_QR_ATTN  = WS_REL_ATTN + (size_t)kVocab*kAttn; // batch x 64 (bias folded)
constexpr size_t WS_HRH      = WS_QR_ATTN + (size_t)kBatch*kAttn;  // vocab x 128 expmap0(rela)
constexpr size_t WS_HRN2     = WS_HRH + (size_t)kVocab*kInDim;     // vocab   ||hr_h||^2
constexpr size_t WS_QNORM    = WS_HRN2 + kVocab;                   // batch   ||qpe||
constexpr size_t WS_ALPHA    = 81600;                              // edge alpha
constexpr size_t WS_PLOG     = WS_ALPHA + kNEdge;                  // edge path logit
constexpr size_t WS_PEXP     = WS_PLOG + kNEdge;                   // edge exp(logit-max)
constexpr size_t WS_PMAX     = WS_PEXP + kNEdge;                   // node max key (uint)
constexpr size_t WS_PSUM     = WS_PMAX + kNNode;                   // node sum exp
constexpr size_t WS_PWSUM    = WS_PSUM + kNNode;                   // node sum pw
constexpr size_t WS_DISPA    = WS_PWSUM + kNNode;                  // node sum pw*||pc||^2
constexpr size_t WS_END      = WS_DISPA + kNNode;
static_assert(WS_ALPHA >= WS_QNORM + kBatch, "ws layout");

// ---------------- output layout (floats, concatenated tuple) ----------------
constexpr size_t OUT_HID   = 0;                                  // 250000x128 (used as agg accumulator first)
constexpr size_t OUT_PSN   = (size_t)kNNode * kInDim;            // 32,000,000
constexpr size_t OUT_DISP  = OUT_PSN + (size_t)kNNode * kDPath;  // 48,000,000
constexpr size_t OUT_NODES = OUT_DISP + kNNode;                  // 48,250,000
constexpr size_t OUT_MASK  = OUT_NODES + (size_t)kNNode * 2;     // 48,750,000

// ---------------- helpers ----------------
__device__ __forceinline__ float sanf(float x) {           // _san: nan->0, inf->+-1e6
  if (isnan(x)) return 0.f;
  if (isinf(x)) return x > 0.f ? 1.0e6f : -1.0e6f;
  return x;
}
__device__ __forceinline__ float tanh_c(float x) { return tanhf(fminf(fmaxf(x, -15.f), 15.f)); }
__device__ __forceinline__ float artanh_(float x) {
  float xc = fminf(fmaxf(x, -1.f + 1e-5f), 1.f - 1e-5f);
  return 0.5f * (log1pf(xc) - log1pf(-xc));
}
__device__ __forceinline__ float sigmoid_(float x) { return 1.f / (1.f + expf(-x)); }
__device__ __forceinline__ float wred(float v) {            // wave32 butterfly sum (all lanes)
#pragma unroll
  for (int m = 16; m >= 1; m >>= 1) v += __shfl_xor(v, m, 32);
  return v;
}
__device__ __forceinline__ void wred2(float& a, float& b) { // fused pair reduction
#pragma unroll
  for (int m = 16; m >= 1; m >>= 1) { a += __shfl_xor(a, m, 32); b += __shfl_xor(b, m, 32); }
}
// order-preserving float->uint key for atomicMax over signed floats
__device__ __forceinline__ unsigned fkey(float f) {
  unsigned u = __float_as_uint(f);
  return (u & 0x80000000u) ? ~u : (u | 0x80000000u);
}
__device__ __forceinline__ float fdec(unsigned k) {
  return (k & 0x80000000u) ? __uint_as_float(k ^ 0x80000000u) : __uint_as_float(~k);
}

// ================= K0: zero agg/psn regions of out + node counters in ws =================
__global__ void k_zero(float* __restrict__ out, float* __restrict__ ws) {
  size_t i = (size_t)blockIdx.x * blockDim.x + threadIdx.x;
  size_t stride = (size_t)gridDim.x * blockDim.x;
  for (size_t j = i; j < OUT_DISP; j += stride) out[j] = 0.f;                  // 48M floats
  for (size_t j = i; j < (WS_END - WS_PMAX); j += stride) ws[WS_PMAX + j] = 0.f; // pmax key 0 == -inf
}

// ================= K1a: rel_attn = rela@Wr ; qr_attn = rela[q_rel]@Wqr + b =================
__global__ void k_pre_attn(const float* __restrict__ rela, const float* __restrict__ Wr,
                           const float* __restrict__ Wqr_w, const float* __restrict__ Wqr_b,
                           const int* __restrict__ q_rel, float* __restrict__ ws) {
  int t = blockIdx.x * blockDim.x + threadIdx.x;
  if (t < kVocab * kAttn) {
    int r = t / kAttn, n = t % kAttn;
    float a = 0.f;
#pragma unroll 8
    for (int k = 0; k < kInDim; ++k) a = fmaf(rela[r * kInDim + k], Wr[k * kAttn + n], a);
    ws[WS_REL_ATTN + t] = a;
  } else if (t < kVocab * kAttn + kBatch * kAttn) {
    int u = t - kVocab * kAttn;
    int b = u / kAttn, n = u % kAttn;
    int rr = q_rel[b];
    float a = Wqr_b[n];
#pragma unroll 8
    for (int k = 0; k < kInDim; ++k) a = fmaf(rela[rr * kInDim + k], Wqr_w[k * kAttn + n], a);
    ws[WS_QR_ATTN + u] = a;
  }
}

// ================= K1b: hr_h = expmap0(rela) per vocab row; qnorm per batch =================
__global__ void k_pre_hyp(const float* __restrict__ rela, const float* __restrict__ qproto,
                          const float* __restrict__ curv, float* __restrict__ ws) {
  int wv = (blockIdx.x * blockDim.x + threadIdx.x) >> 5;
  int lane = threadIdx.x & 31;
  float c = fmaxf(curv[0], 1e-6f), sc = sqrtf(c), maxn = (1.f - 0.004f) / sc;
  if (wv < kVocab) {
    const float* u = rela + (size_t)wv * kInDim;
    float x[4], n2 = 0.f;
#pragma unroll
    for (int j = 0; j < 4; ++j) { x[j] = u[lane * 4 + j]; n2 = fmaf(x[j], x[j], n2); }
    n2 = wred(n2);
    float un = fmaxf(sqrtf(n2), 1e-15f);
    float g = tanh_c(sc * un) / (sc * un);        // expmap0 scale
    float gn = un * g;
    float pf = (gn > maxn) ? maxn / fmaxf(gn, 1e-15f) : 1.f;  // project
    float hn2 = 0.f;
    float* hr = ws + WS_HRH + (size_t)wv * kInDim;
#pragma unroll
    for (int j = 0; j < 4; ++j) {
      float v = sanf(x[j] * g * pf);
      hr[lane * 4 + j] = v;
      hn2 = fmaf(v, v, hn2);
    }
    hn2 = wred(hn2);
    if (lane == 0) ws[WS_HRN2 + wv] = hn2;
  } else if (wv < kVocab + kBatch) {
    int b = wv - kVocab;
    float q0 = qproto[b * kDPath + lane * 2], q1 = qproto[b * kDPath + lane * 2 + 1];
    float n2 = wred(fmaf(q0, q0, q1 * q1));
    if (lane == 0) ws[WS_QNORM + b] = fmaxf(sqrtf(n2), 1e-8f);
  }
}

// ================= K2: attention alpha via bf16 WMMA =================
// block = 128 thr (4 waves); each wave owns 16 edges: attn16x64 = hs16x128 @ Ws128x64
__global__ void __launch_bounds__(128)
k_edge_attn(const float* __restrict__ hidden, const int* __restrict__ edges,
            const float* __restrict__ Ws, const float* __restrict__ w_al,
            const float* __restrict__ w_al_b, float* __restrict__ ws) {
  __shared__ __bf16 sWsT[kAttn * kInDim];  // Ws transposed [n][k], bf16, 16KB
  for (int idx = threadIdx.x; idx < kAttn * kInDim; idx += blockDim.x) {
    int n = idx >> 7, k = idx & 127;
    sWsT[idx] = (__bf16)Ws[k * kAttn + n];
  }
  __syncthreads();

  const int w = threadIdx.x >> 5, lane = threadIdx.x & 31;
  const int hi = lane >> 4, lo = lane & 15;
  const long eb = (long)blockIdx.x * 64 + (long)w * 16;
  if (eb >= kNEdge) return;

  // A row for this lane: edge eb+lo (16-bit A: lanes0-15 and 16-31 both map M=lane&15)
  long eA = eb + lo; if (eA >= kNEdge) eA = kNEdge - 1;
  const int sub = edges[eA * 6 + 4];
  const float* hsrow = hidden + (size_t)sub * kInDim;

  v8f cacc[4] = {};   // 4 N-tiles of 16x16 f32
#pragma unroll
  for (int kc = 0; kc < 4; ++kc) {          // K = 128 in 32-chunks
    const int r1 = kc * 32 + hi * 8;        // lane's first K run
    const int r2 = r1 + 16;                 // lane's second K run
    v16bf a;
#pragma unroll
    for (int j = 0; j < 8; ++j) {
      a[j]     = (__bf16)hsrow[r1 + j];
      a[8 + j] = (__bf16)hsrow[r2 + j];
    }
    if (kc < 3) __builtin_prefetch(hsrow + r1 + 32, 0, 1);  // global_prefetch_b8
#pragma unroll
    for (int t = 0; t < 4; ++t) {
      const __bf16* wrow = &sWsT[(t * 16 + lo) * kInDim];
      v16bf b;
#pragma unroll
      for (int j = 0; j < 8; ++j) { b[j] = wrow[r1 + j]; b[8 + j] = wrow[r2 + j]; }
      cacc[t] = __builtin_amdgcn_wmma_f32_16x16x32_bf16(
          false, a, false, b, (short)0, cacc[t], false, false);
    }
  }

  // epilogue: + rel_attn + qr_attn, relu, dot w_alpha, sigmoid
  const float* rel_attn = ws + WS_REL_ATTN;
  const float* qr_attn  = ws + WS_QR_ATTN;
  float wa[4];
#pragma unroll
  for (int t = 0; t < 4; ++t) wa[t] = w_al[t * 16 + lo];
  int relv[8], riv[8];
#pragma unroll
  for (int i = 0; i < 8; ++i) {
    long e = eb + hi * 8 + i; if (e >= kNEdge) e = kNEdge - 1;
    relv[i] = edges[e * 6 + 2];
    riv[i]  = edges[e * 6 + 0];
  }
  float acc[8] = {};
#pragma unroll
  for (int t = 0; t < 4; ++t) {
    const int N = t * 16 + lo;
#pragma unroll
    for (int i = 0; i < 8; ++i) {
      float v = cacc[t][i] + rel_attn[relv[i] * kAttn + N] + qr_attn[riv[i] * kAttn + N];
      v = fmaxf(v, 0.f);
      acc[i] = fmaf(v, wa[t], acc[i]);
    }
  }
#pragma unroll
  for (int m = 1; m <= 8; m <<= 1)
#pragma unroll
    for (int i = 0; i < 8; ++i) acc[i] += __shfl_xor(acc[i], m, 32);
  if (lo == 0) {
    const float bAl = w_al_b[0];
#pragma unroll
    for (int i = 0; i < 8; ++i) {
      long e = eb + hi * 8 + i;
      if (e < kNEdge) ws[WS_ALPHA + e] = sigmoid_(acc[i] + bAl);
    }
  }
}

// ================= K3: hyperbolic message + scatter; path logits + segment max =================
// one wave (32 lanes) per edge; 4 hidden elems / lane, 2 path elems / lane
__global__ void __launch_bounds__(256)
k_edge_msg(const float* __restrict__ hidden, const float* __restrict__ pstate,
           const int* __restrict__ edges, const float* __restrict__ prel,
           const float* __restrict__ qproto, const float* __restrict__ curv,
           const float* __restrict__ lam, float* __restrict__ ws, float* __restrict__ out) {
  const long e = (long)blockIdx.x * (blockDim.x >> 5) + (threadIdx.x >> 5);
  if (e >= kNEdge) return;
  const int lane = threadIdx.x & 31;
  const float c = fmaxf(curv[0], 1e-6f), sc = sqrtf(c), maxn = (1.f - 0.004f) / sc;

  const int ridx = edges[e * 6 + 0];
  const int rel  = edges[e * 6 + 2];
  const int sub  = edges[e * 6 + 4];
  const int obj  = edges[e * 6 + 5];
  const bool selfloop = (rel == 2 * kNRel);
  const float alpha_e = ws[WS_ALPHA + e];

  // hs_h = project(expmap0(hs))
  const float* hsrow = hidden + (size_t)sub * kInDim;
  float x[4], n2 = 0.f;
#pragma unroll
  for (int j = 0; j < 4; ++j) { x[j] = hsrow[lane * 4 + j]; n2 = fmaf(x[j], x[j], n2); }
  n2 = wred(n2);
  float un = fmaxf(sqrtf(n2), 1e-15f);
  float g = tanh_c(sc * un) / (sc * un);
  float gn = un * g;
  float pf = (gn > maxn) ? maxn / fmaxf(gn, 1e-15f) : 1.f;
  float xh[4];
#pragma unroll
  for (int j = 0; j < 4; ++j) xh[j] = sanf(x[j] * g * pf);

  // mobius_add(hs_h, hr_h) with precomputed hr_h / ||hr_h||^2
  const float* hr = ws + WS_HRH + (size_t)rel * kInDim;
  const float y2 = ws[WS_HRN2 + rel];
  float y[4], x2 = 0.f, xy = 0.f;
#pragma unroll
  for (int j = 0; j < 4; ++j) {
    y[j] = hr[lane * 4 + j];
    x2 = fmaf(xh[j], xh[j], x2);
    xy = fmaf(xh[j], y[j], xy);
  }
  wred2(x2, xy);
  const float kx = 1.f + 2.f * c * xy + c * y2;
  const float ky = 1.f - c * x2;
  const float den = fmaxf(1.f + 2.f * c * xy + c * c * x2 * y2, 1e-15f);
  float mvec[4], m2 = 0.f;
#pragma unroll
  for (int j = 0; j < 4; ++j) {
    float v = sanf((kx * xh[j] + ky * y[j]) / den);
    mvec[j] = v; m2 = fmaf(v, v, m2);
  }
  m2 = wred(m2);
  float mn = fmaxf(sqrtf(m2), 1e-15f);
  float pf2 = (mn > maxn) ? maxn / mn : 1.f;   // project (idempotent; ref projects twice)
  float mnp = fminf(mn, maxn);
  // logmap0
  float lf = artanh_(fminf(sc * mnp, 1.f - 0.004f)) / (sc * fmaxf(mnp, 1e-15f));
  float* agg = out + OUT_HID;
#pragma unroll
  for (int j = 0; j < 4; ++j) {
    float msg = alpha_e * sanf(mvec[j] * pf2 * lf);
    atomicAdd(&agg[(size_t)obj * kInDim + lane * 4 + j], msg);
  }

  // ---- path logits ----
  float pc0 = sanf(pstate[(size_t)sub * kDPath + lane * 2 + 0] + (selfloop ? 0.f : prel[(size_t)e * kDPath + lane * 2 + 0]));
  float pc1 = sanf(pstate[(size_t)sub * kDPath + lane * 2 + 1] + (selfloop ? 0.f : prel[(size_t)e * kDPath + lane * 2 + 1]));
  float q0 = qproto[ridx * kDPath + lane * 2 + 0];
  float q1 = qproto[ridx * kDPath + lane * 2 + 1];
  float cn2 = fmaf(pc0, pc0, pc1 * pc1);
  float dot = fmaf(pc0, q0, pc1 * q1);
  wred2(cn2, dot);
  float cn = fmaxf(sqrtf(cn2), 1e-8f);
  float qn = ws[WS_QNORM + ridx];
  float cosv = dot / (cn * qn);
  float pl = logf(fmaxf(alpha_e, 1e-15f)) + lam[0] * cosv;
  if (isnan(pl)) pl = -50.f;
  pl = fminf(fmaxf(pl, -50.f), 50.f);
  if (lane == 0) {
    ws[WS_PLOG + e] = pl;
    atomicMax((unsigned*)(ws + WS_PMAX) + obj, fkey(pl));
  }
}

// ================= K4: pexp = exp(logit - segmax); psum += =================
__global__ void k_edge_exp(const int* __restrict__ edges, float* __restrict__ ws) {
  long e = (long)blockIdx.x * blockDim.x + threadIdx.x;
  if (e >= kNEdge) return;
  const int obj = edges[e * 6 + 5];
  const float m = fdec(((const unsigned*)(ws + WS_PMAX))[obj]);
  float pe = expf(ws[WS_PLOG + e] - m);
  if (!isfinite(pe)) pe = 0.f;
  ws[WS_PEXP + e] = pe;
  atomicAdd(&ws[WS_PSUM + obj], pe);
}

// ================= K5: weighted path scatter (psn, Sum pw, Sum pw*||pc||^2) =================
__global__ void __launch_bounds__(256)
k_edge_path(const float* __restrict__ pstate, const int* __restrict__ edges,
            const float* __restrict__ prel, float* __restrict__ ws, float* __restrict__ out) {
  const long e = (long)blockIdx.x * (blockDim.x >> 5) + (threadIdx.x >> 5);
  if (e >= kNEdge) return;
  const int lane = threadIdx.x & 31;
  const int rel = edges[e * 6 + 2];
  const int sub = edges[e * 6 + 4];
  const int obj = edges[e * 6 + 5];
  const bool selfloop = (rel == 2 * kNRel);
  float pw = ws[WS_PEXP + e] / fmaxf(ws[WS_PSUM + obj], 1e-15f);
  if (!isfinite(pw)) pw = 0.f;
  float pc0 = sanf(pstate[(size_t)sub * kDPath + lane * 2 + 0] + (selfloop ? 0.f : prel[(size_t)e * kDPath + lane * 2 + 0]));
  float pc1 = sanf(pstate[(size_t)sub * kDPath + lane * 2 + 1] + (selfloop ? 0.f : prel[(size_t)e * kDPath + lane * 2 + 1]));
  float* psn = out + OUT_PSN + (size_t)obj * kDPath;
  atomicAdd(&psn[lane * 2 + 0], pw * pc0);
  atomicAdd(&psn[lane * 2 + 1], pw * pc1);
  float cn2 = wred(fmaf(pc0, pc0, pc1 * pc1));
  if (lane == 0) {
    atomicAdd(&ws[WS_DISPA + obj], pw * cn2);
    atomicAdd(&ws[WS_PWSUM + obj], pw);
  }
}

// ================= K6: hidden_new = logmap0(expmap0(agg @ W_h)) via f32 WMMA =================
// block = 128 thr (4 waves); each wave: 16 nodes x 128 out, K=128 in 4-steps (in-place over agg)
__global__ void __launch_bounds__(128)
k_node_gemm(const float* __restrict__ W_h, const float* __restrict__ curv, float* __restrict__ out) {
  __shared__ float sW[kInDim * 128];  // W_h row-major [k][n], 64KB
  for (int idx = threadIdx.x; idx < kInDim * 128; idx += blockDim.x) sW[idx] = W_h[idx];
  __syncthreads();

  const int w = threadIdx.x >> 5, lane = threadIdx.x & 31;
  const int hi = lane >> 4, lo = lane & 15;
  const long nb = (long)blockIdx.x * 64 + (long)w * 16;
  if (nb >= kNNode) return;

  long nrow = nb + lo; if (nrow >= kNNode) nrow = kNNode - 1;
  const float* arow = out + OUT_HID + (size_t)nrow * kInDim;

  v8f cacc[8] = {};  // 8 N-tiles (128 cols)
#pragma unroll 4
  for (int ks = 0; ks < 32; ++ks) {
    const int k0 = ks * 4 + hi * 2;      // f32 A 16x4: lane holds K {k0,k0+1}
    v2f a; a.x = arow[k0]; a.y = arow[k0 + 1];
#pragma unroll
    for (int nt = 0; nt < 8; ++nt) {
      const int N = nt * 16 + lo;
      v2f b; b.x = sW[k0 * 128 + N]; b.y = sW[(k0 + 1) * 128 + N];
      cacc[nt] = __builtin_amdgcn_wmma_f32_16x16x4_f32(
          false, a, false, b, (short)0, cacc[nt], false, false);
    }
  }

  // per-row logmap0(project(expmap0(a))) scale factor, then in-place store
  const float c = fmaxf(curv[0], 1e-6f), sc = sqrtf(c), maxn = (1.f - 0.004f) / sc;
  float* hid = out + OUT_HID;
#pragma unroll
  for (int i = 0; i < 8; ++i) {
    float ss = 0.f;
#pragma unroll
    for (int nt = 0; nt < 8; ++nt) ss = fmaf(cacc[nt][i], cacc[nt][i], ss);
#pragma unroll
    for (int m = 1; m <= 8; m <<= 1) ss += __shfl_xor(ss, m, 32);  // over the 16-lane half
    float un = fmaxf(sqrtf(ss), 1e-15f);
    float g = tanh_c(sc * un) / (sc * un);
    float gn = un * g;
    float pf = (gn > maxn) ? maxn / fmaxf(gn, 1e-15f) : 1.f;
    float yn = fminf(gn, maxn);
    float lf = artanh_(fminf(sc * yn, 1.f - 0.004f)) / (sc * fmaxf(yn, 1e-15f));
    float f = g * pf * lf;
    long n = nb + hi * 8 + i;  // C rows: M = i + 8*hi
    if (n < kNNode) {
#pragma unroll
      for (int nt = 0; nt < 8; ++nt) hid[(size_t)n * kInDim + nt * 16 + lo] = sanf(cacc[nt][i] * f);
    }
  }
}

// ================= K7: finalize psn/disp/nodes/mask =================
// disp = Sum pw||pc||^2 - (2 - Sum pw) * ||psn||^2   (exact identity, same accumulators)
__global__ void k_node_fin(const int* __restrict__ nodes, float* __restrict__ ws, float* __restrict__ out) {
  long n = (long)blockIdx.x * blockDim.x + threadIdx.x;
  if (n >= kNNode) return;
  float S = ws[WS_PWSUM + n];
  float A = ws[WS_DISPA + n];
  float* psn = out + OUT_PSN + (size_t)n * kDPath;
  float nn2 = 0.f;
#pragma unroll 8
  for (int j = 0; j < kDPath; ++j) {
    float p = psn[j];
    nn2 = fmaf(p, p, nn2);
    psn[j] = sanf(p);
  }
  float disp = A - (2.f - S) * nn2;
  if (isnan(disp)) disp = 0.f;
  disp = fminf(fmaxf(disp, 0.f), 1.0e6f);
  out[OUT_DISP + n] = disp;
  out[OUT_NODES + 2 * n + 0] = (float)nodes[2 * n + 0];
  out[OUT_NODES + 2 * n + 1] = (float)nodes[2 * n + 1];
  out[OUT_MASK + n] = 1.0f;
}

// ================= launcher =================
extern "C" void kernel_launch(void* const* d_in, const int* in_sizes, int n_in,
                              void* d_out, int out_size, void* d_ws, size_t ws_size,
                              hipStream_t stream) {
  const int*   q_rel  = (const int*)  d_in[1];
  const float* hidden = (const float*)d_in[2];
  const float* pstate = (const float*)d_in[3];
  const int*   edges  = (const int*)  d_in[4];
  const int*   nodes  = (const int*)  d_in[5];
  const float* curv   = (const float*)d_in[8];
  const float* prel   = (const float*)d_in[9];
  const float* qproto = (const float*)d_in[10];
  const float* lam    = (const float*)d_in[11];
  const float* rela   = (const float*)d_in[12];
  const float* Ws     = (const float*)d_in[13];
  const float* Wr     = (const float*)d_in[14];
  const float* Wqr_w  = (const float*)d_in[15];
  const float* Wqr_b  = (const float*)d_in[16];
  const float* w_al   = (const float*)d_in[17];
  const float* w_al_b = (const float*)d_in[18];
  const float* W_h    = (const float*)d_in[19];
  float* out = (float*)d_out;
  float* ws  = (float*)d_ws;

  k_zero     <<<2048, 256, 0, stream>>>(out, ws);
  k_pre_attn <<<117, 256, 0, stream>>>(rela, Wr, Wqr_w, Wqr_b, q_rel, ws);
  k_pre_hyp  <<<117, 128, 0, stream>>>(rela, qproto, curv, ws);
  k_edge_attn<<<(kNEdge + 63) / 64, 128, 0, stream>>>(hidden, edges, Ws, w_al, w_al_b, ws);
  k_edge_msg <<<(kNEdge + 7) / 8, 256, 0, stream>>>(hidden, pstate, edges, prel, qproto, curv, lam, ws, out);
  k_edge_exp <<<(kNEdge + 255) / 256, 256, 0, stream>>>(edges, ws);
  k_edge_path<<<(kNEdge + 7) / 8, 256, 0, stream>>>(pstate, edges, prel, ws, out);
  k_node_gemm<<<(kNNode + 63) / 64, 128, 0, stream>>>(W_h, curv, out);
  k_node_fin <<<(kNNode + 255) / 256, 256, 0, stream>>>(nodes, ws, out);
}
